// MultiHeadAttention_43688407335116
// MI455X (gfx1250) — matrix operand order repaired and checked by reading, not compiled, not run on hardware
//
#include <hip/hip_runtime.h>
#include <math.h>

// ---------------- problem constants (from reference) ----------------
#define BB    2
#define NN    2048
#define DD    768
#define HH    12
#define DH    64
#define MM    4096          // BB*NN rows
#define QKVN  2304          // 3*HH*DH
#define DP    192
#define VSTRIDE 4096        // padded row stride of transposed bias table

typedef __attribute__((ext_vector_type(16))) _Float16 v16h;
typedef __attribute__((ext_vector_type(8)))  _Float16 h8;
typedef __attribute__((ext_vector_type(8)))  float    v8f;

union hv16 { v16h v; h8 h[2]; };

__device__ __forceinline__ v8f wmma_f16(v16h a, v16h b, v8f c) {
  // D = A(16x32 f16) * B(32x16 f16) + C(16x16 f32)
  return __builtin_amdgcn_wmma_f32_16x16x32_f16(
      /*neg_a=*/false, a, /*neg_b=*/false, b,
      /*c_mod=*/(short)0, c, /*reuse_a=*/false, /*reuse_b=*/false);
}

__device__ __forceinline__ v8f zero8() {
  v8f z;
#pragma unroll
  for (int i = 0; i < 8; ++i) z[i] = 0.0f;
  return z;
}

// ---------------- fp32 -> fp16 conversion ----------------
__global__ void f32_to_f16_kernel(const float* __restrict__ src,
                                  _Float16* __restrict__ dst, int n) {
  int i = blockIdx.x * blockDim.x + threadIdx.x;
  if (i < n) dst[i] = (_Float16)src[i];
}

// ---------------- DPB MLP -> transposed table valsT[H][VSTRIDE] ----------
__device__ __forceinline__ float block_sum_192(float v, float* red) {
#pragma unroll
  for (int m = 16; m >= 1; m >>= 1) v += __shfl_xor(v, m);
  int wid = threadIdx.x >> 5;
  if ((threadIdx.x & 31) == 0) red[wid] = v;
  __syncthreads();
  float s = red[0] + red[1] + red[2] + red[3] + red[4] + red[5];
  __syncthreads();
  return s;
}

__global__ __launch_bounds__(DP) void dpb_kernel(
    const float* __restrict__ w_in,  const float* __restrict__ b_in,
    const float* __restrict__ w_hid, const float* __restrict__ b_hid,
    const float* __restrict__ ln_g,  const float* __restrict__ ln_b,
    const float* __restrict__ w_out, const float* __restrict__ b_out,
    float* __restrict__ valsT) {
  __shared__ float hbuf[DP];
  __shared__ float red[8];
  const int j = threadIdx.x;
  const int r = blockIdx.x;                       // 0 .. 2N-2
  float pos = (float)(r - (NN - 1));
  float p   = copysignf(log1pf(fabsf(pos)), pos); // sign(x)*log(|x|+1)

  float hv = p * w_in[j] + b_in[j];
  for (int l = 0; l < 4; ++l) {
    if (l > 0) {
      const float* W = w_hid + (l - 1) * DP * DP;
      float acc = b_hid[(l - 1) * DP + j];
      for (int i = 0; i < DP; ++i) acc += hbuf[i] * W[i * DP + j];
      __syncthreads();
      hv = acc;
    }
    float S1   = block_sum_192(hv, red);
    float S2   = block_sum_192(hv * hv, red);
    float mean = S1 * (1.0f / DP);
    float var  = S2 * (1.0f / DP) - mean * mean;
    float xn   = (hv - mean) * rsqrtf(var + 1e-5f);
    float y    = xn * ln_g[l * DP + j] + ln_b[l * DP + j];
    hv = y / (1.0f + __expf(-y));                 // silu
    __syncthreads();
    hbuf[j] = hv;
    __syncthreads();
  }
  if (j < HH) {
    float acc = b_out[j];
    for (int i = 0; i < DP; ++i) acc += hbuf[i] * w_out[i * HH + j];
    valsT[j * VSTRIDE + r] = acc;                 // transposed: [head][rel-pos]
  }
}

// ---------------- tiled WMMA GEMM: C[M,NCOLS] = A[M,K] * W[K,NCOLS] + bias ---
// Block 256 threads = 8 waves in 4(M) x 2(N); tile 128 x 128; K step 32.
// Double-buffered LDS: one barrier per K step; 8 WMMAs/wave/step.
template <int NCOLS, int K>
__device__ __forceinline__ void stage_tiles(
    const _Float16* __restrict__ A, const _Float16* __restrict__ W,
    int m0, int n0, int kt, int tid,
    _Float16 (*__restrict__ As)[32], _Float16 (*__restrict__ Bs)[32]) {
  // A tile: 128x32 halves, two coalesced 16B loads per thread
  int row = tid >> 1;
  int col = (tid & 1) * 16;
  const _Float16* src = &A[(m0 + row) * K + kt + col];
  *reinterpret_cast<h8*>(&As[row][col])     = *reinterpret_cast<const h8*>(src);
  *reinterpret_cast<h8*>(&As[row][col + 8]) = *reinterpret_cast<const h8*>(src + 8);
  // W tile transposed into Bs[n][k]
#pragma unroll
  for (int it = 0; it < 2; ++it) {
    int kk = (tid >> 4) + it * 16;  // 0..31
    int nc = (tid & 15) * 8;
    h8 v = *reinterpret_cast<const h8*>(&W[(kt + kk) * NCOLS + n0 + nc]);
#pragma unroll
    for (int e = 0; e < 8; ++e) Bs[nc + e][kk] = v[e];
  }
}

template <int NCOLS, int K, bool OUT_F32>
__global__ __launch_bounds__(256) void gemm_f16_kernel(
    const _Float16* __restrict__ A, const _Float16* __restrict__ W,
    const float* __restrict__ bias, void* __restrict__ Cout) {
  __shared__ __align__(16) _Float16 As[2][128][32];
  __shared__ __align__(16) _Float16 Bs[2][128][32];
  const int tid  = threadIdx.x;
  const int lane = tid & 31;
  const int w    = tid >> 5;
  const int wm   = w >> 1;          // 0..3  (32 rows each)
  const int wn   = w & 1;           // 0..1  (64 cols each)
  const int m0   = blockIdx.y * 128;
  const int n0   = blockIdx.x * 128;
  const int l16  = lane & 15;
  const int hi   = lane >> 4;       // 0/1 wave half

  v8f acc[2][4];
#pragma unroll
  for (int i = 0; i < 2; ++i)
#pragma unroll
    for (int jn = 0; jn < 4; ++jn) acc[i][jn] = zero8();

  stage_tiles<NCOLS, K>(A, W, m0, n0, 0, tid, As[0], Bs[0]);
  int buf = 0;
  for (int kt = 0; kt < K; kt += 32) {
    __syncthreads();                 // staged tile visible; prior reads done
    if (kt + 32 < K)                 // prefetch next tile into other buffer
      stage_tiles<NCOLS, K>(A, W, m0, n0, kt + 32, tid, As[buf ^ 1], Bs[buf ^ 1]);

    v16h a[2], b[4];
#pragma unroll
    for (int mf = 0; mf < 2; ++mf) { // A frag: lane holds row m, k pattern
      int m  = wm * 32 + mf * 16 + l16;
      int kb = hi * 16;
      hv16 t;
      t.h[0] = *reinterpret_cast<const h8*>(&As[buf][m][kb]);
      t.h[1] = *reinterpret_cast<const h8*>(&As[buf][m][kb + 8]);
      a[mf] = t.v;
    }
#pragma unroll
    for (int nf = 0; nf < 4; ++nf) { // B frag: lane holds col n, contiguous k
      int n  = wn * 64 + nf * 16 + l16;
      int kb = hi * 16;
      hv16 t;
      t.h[0] = *reinterpret_cast<const h8*>(&Bs[buf][n][kb]);
      t.h[1] = *reinterpret_cast<const h8*>(&Bs[buf][n][kb + 8]);
      b[nf] = t.v;
    }
#pragma unroll
    for (int mf = 0; mf < 2; ++mf)
#pragma unroll
      for (int nf = 0; nf < 4; ++nf)
        acc[mf][nf] = wmma_f16(a[mf], b[nf], acc[mf][nf]);
    buf ^= 1;
  }

  float*    Cf = (float*)Cout;
  _Float16* Ch = (_Float16*)Cout;
#pragma unroll
  for (int mf = 0; mf < 2; ++mf)
#pragma unroll
    for (int nf = 0; nf < 4; ++nf) {
      int n    = n0 + wn * 64 + nf * 16 + l16;
      float bv = bias[n];
#pragma unroll
      for (int r = 0; r < 8; ++r) {
        int m   = m0 + wm * 32 + mf * 16 + r + hi * 8;
        float v = acc[mf][nf][r] + bv;
        if constexpr (OUT_F32) Cf[m * NCOLS + n] = v;
        else                   Ch[m * NCOLS + n] = (_Float16)v;
      }
    }
}

// ---------------- reshape: qkv[4096,2304] -> Q/K [b,h,n,d], Vt [b,h,d,n] ----
__global__ void reshape_qkv_kernel(const _Float16* __restrict__ qkv,
                                   _Float16* __restrict__ Q,
                                   _Float16* __restrict__ Kd,
                                   _Float16* __restrict__ Vt) {
  size_t idx = (size_t)blockIdx.x * blockDim.x + threadIdx.x; // < 4096*2304
  int col = (int)(idx % QKVN);
  int row = (int)(idx / QKVN);
  int b = row >> 11, n = row & 2047;
  int part = col / (HH * DH);
  int c = col % (HH * DH);
  int hh = c >> 6, d = c & 63;
  _Float16 v = qkv[idx];
  int bh = b * HH + hh;
  if (part == 0)      Q [(bh * NN + n) * DH + d] = v;
  else if (part == 1) Kd[(bh * NN + n) * DH + d] = v;
  else                Vt[(bh * DH + d) * NN + n] = v;
}

// ---------------- flash attention with WMMA + DPB bias ----------------
// grid (NN/128, BB*HH); 8 waves/block; wave owns 16 query rows.
__global__ __launch_bounds__(256) void attn_kernel(
    const _Float16* __restrict__ Q, const _Float16* __restrict__ Kd,
    const _Float16* __restrict__ Vt, const float* __restrict__ valsT,
    _Float16* __restrict__ Oh) {
  __shared__ __align__(16) _Float16 P[8][16][32];  // per-wave P staging
  const int tid = threadIdx.x, lane = tid & 31, w = tid >> 5;
  const int l16 = lane & 15;
  const int hi  = lane >> 4;
  const int bh  = blockIdx.y;           // 0..23
  const int b   = bh / HH, h = bh % HH;
  const int i0  = blockIdx.x * 128 + w * 16;
  const _Float16* Qp = Q  + bh * NN * DH + i0 * DH;
  const _Float16* Kp = Kd + bh * NN * DH;
  const _Float16* Vp = Vt + bh * NN * DH;   // [DH][NN]
  const float*    Vb = valsT + h * VSTRIDE; // bias row for this head

  // Q fragments (resident): k 0..31 and 32..63
  v16h q0, q1;
  {
    int kb = hi * 16;
    hv16 t;
    t.h[0] = *(const h8*)&Qp[l16 * DH + kb];
    t.h[1] = *(const h8*)&Qp[l16 * DH + kb + 8];
    q0 = t.v;
    t.h[0] = *(const h8*)&Qp[l16 * DH + 32 + kb];
    t.h[1] = *(const h8*)&Qp[l16 * DH + 32 + kb + 8];
    q1 = t.v;
  }

  v8f oacc[4];
#pragma unroll
  for (int nf = 0; nf < 4; ++nf) oacc[nf] = zero8();
  float mrow[8], lrow[8];
#pragma unroll
  for (int r = 0; r < 8; ++r) { mrow[r] = -3.0e38f; lrow[r] = 0.0f; }

  for (int jj = 0; jj < NN; jj += 32) {
    if (jj + 32 < NN) __builtin_prefetch(&Kp[(jj + 32 + l16) * DH], 0, 0);
    v8f s01[2];
#pragma unroll
    for (int jt = 0; jt < 2; ++jt) {
      int j0 = jj + jt * 16;
      int j  = j0 + l16;
      int kb = hi * 16;
      hv16 t;
      v8f s = zero8();
      t.h[0] = *(const h8*)&Kp[j * DH + kb];
      t.h[1] = *(const h8*)&Kp[j * DH + kb + 8];
      s = wmma_f16(q0, t.v, s);
      t.h[0] = *(const h8*)&Kp[j * DH + 32 + kb];
      t.h[1] = *(const h8*)&Kp[j * DH + 32 + kb + 8];
      s = wmma_f16(q1, t.v, s);
      // scale + DPB bias. Accumulator layout: VGPR r -> row i0+r+hi*8,
      // lane -> col j0+l16. Transposed table => 8 consecutive floats/lane.
      const float* bp = Vb + (i0 + hi * 8 + (NN - 1) - j);
#pragma unroll
      for (int r = 0; r < 8; ++r)
        s[r] = s[r] * 0.125f + bp[r];
      s01[jt] = s;
    }
    // online softmax: row stats live per (VGPR r, wave-half)
#pragma unroll
    for (int r = 0; r < 8; ++r) {
      float mx = fmaxf(s01[0][r], s01[1][r]);
      mx = fmaxf(mx, __shfl_xor(mx, 1));
      mx = fmaxf(mx, __shfl_xor(mx, 2));
      mx = fmaxf(mx, __shfl_xor(mx, 4));
      mx = fmaxf(mx, __shfl_xor(mx, 8));
      float mnew = fmaxf(mrow[r], mx);
      float corr = __expf(mrow[r] - mnew);
      float p0   = __expf(s01[0][r] - mnew);
      float p1   = __expf(s01[1][r] - mnew);
      float rs   = p0 + p1;
      rs += __shfl_xor(rs, 1);
      rs += __shfl_xor(rs, 2);
      rs += __shfl_xor(rs, 4);
      rs += __shfl_xor(rs, 8);
      lrow[r] = lrow[r] * corr + rs;
      mrow[r] = mnew;
#pragma unroll
      for (int nf = 0; nf < 4; ++nf) oacc[nf][r] *= corr;
      int m = r + hi * 8;
      P[w][m][l16]      = (_Float16)p0;
      P[w][m][16 + l16] = (_Float16)p1;
    }
    asm volatile("s_wait_dscnt 0" ::: "memory"); // wave-local LDS RAW fence
    // P as A-fragment (16x32)
    v16h pa;
    {
      int kb = hi * 16;
      hv16 t;
      t.h[0] = *(const h8*)&P[w][l16][kb];
      t.h[1] = *(const h8*)&P[w][l16][kb + 8];
      pa = t.v;
    }
    // V B-fragments from transposed V: contiguous along keys
#pragma unroll
    for (int nf = 0; nf < 4; ++nf) {
      int d  = nf * 16 + l16;
      int kb = hi * 16;
      hv16 t;
      t.h[0] = *(const h8*)&Vp[d * NN + jj + kb];
      t.h[1] = *(const h8*)&Vp[d * NN + jj + kb + 8];
      oacc[nf] = wmma_f16(pa, t.v, oacc[nf]);
    }
  }
  // epilogue: normalize and scatter to o[b, n, h*64+d] (f16)
#pragma unroll
  for (int r = 0; r < 8; ++r) {
    float inv = 1.0f / lrow[r];
    int row   = i0 + r + hi * 8;
    int ob    = (b * NN + row) * DD + h * DH;
#pragma unroll
    for (int nf = 0; nf < 4; ++nf)
      Oh[ob + nf * 16 + l16] = (_Float16)(oacc[nf][r] * inv);
  }
}

// ---------------- workspace layout (bytes, 256-aligned) ----------------
#define OFF_XH     ((size_t)0)                       // 4096*768 f16
#define OFF_QKVWH  (OFF_XH    + (size_t)MM * DD * 2) // 768*2304 f16
#define OFF_OUTWH  (OFF_QKVWH + (size_t)DD * QKVN * 2)
#define OFF_QKVH   (OFF_OUTWH + (size_t)DD * DD * 2) // 4096*2304 f16
#define OFF_QH     (OFF_QKVH  + (size_t)MM * QKVN * 2)
#define OFF_KH     (OFF_QH    + (size_t)BB * HH * NN * DH * 2)
#define OFF_VTH    (OFF_KH    + (size_t)BB * HH * NN * DH * 2)
#define OFF_OH     (OFF_VTH   + (size_t)BB * HH * NN * DH * 2)
#define OFF_VALS   (OFF_OH    + (size_t)MM * DD * 2) // HH*VSTRIDE f32

extern "C" void kernel_launch(void* const* d_in, const int* in_sizes, int n_in,
                              void* d_out, int out_size, void* d_ws, size_t ws_size,
                              hipStream_t stream) {
  (void)in_sizes; (void)n_in; (void)out_size; (void)ws_size;
  const float* x         = (const float*)d_in[0];
  const float* qkv_w     = (const float*)d_in[1];
  const float* qkv_b     = (const float*)d_in[2];
  const float* out_w     = (const float*)d_in[3];
  const float* out_b     = (const float*)d_in[4];
  const float* dpb_w_in  = (const float*)d_in[5];
  const float* dpb_b_in  = (const float*)d_in[6];
  const float* dpb_w_hid = (const float*)d_in[7];
  const float* dpb_b_hid = (const float*)d_in[8];
  const float* dpb_ln_g  = (const float*)d_in[9];
  const float* dpb_ln_b  = (const float*)d_in[10];
  const float* dpb_w_out = (const float*)d_in[11];
  const float* dpb_b_out = (const float*)d_in[12];

  char* ws = (char*)d_ws;
  _Float16* xh     = (_Float16*)(ws + OFF_XH);
  _Float16* qkvwh  = (_Float16*)(ws + OFF_QKVWH);
  _Float16* outwh  = (_Float16*)(ws + OFF_OUTWH);
  _Float16* qkvh   = (_Float16*)(ws + OFF_QKVH);
  _Float16* Qh     = (_Float16*)(ws + OFF_QH);
  _Float16* Kh     = (_Float16*)(ws + OFF_KH);
  _Float16* Vth    = (_Float16*)(ws + OFF_VTH);
  _Float16* oh     = (_Float16*)(ws + OFF_OH);
  float*    valsT  = (float*)(ws + OFF_VALS);

  // 1) fp32 -> fp16 operands
  f32_to_f16_kernel<<<(MM * DD + 255) / 256, 256, 0, stream>>>(x, xh, MM * DD);
  f32_to_f16_kernel<<<(DD * QKVN + 255) / 256, 256, 0, stream>>>(qkv_w, qkvwh, DD * QKVN);
  f32_to_f16_kernel<<<(DD * DD + 255) / 256, 256, 0, stream>>>(out_w, outwh, DD * DD);

  // 2) dynamic position bias table (transposed) valsT[H][VSTRIDE]
  dpb_kernel<<<2 * NN - 1, DP, 0, stream>>>(
      dpb_w_in, dpb_b_in, dpb_w_hid, dpb_b_hid,
      dpb_ln_g, dpb_ln_b, dpb_w_out, dpb_b_out, valsT);

  // 3) QKV projection (WMMA): [4096,768] x [768,2304] + b -> f16
  gemm_f16_kernel<QKVN, DD, false><<<dim3(QKVN / 128, MM / 128), 256, 0, stream>>>(
      xh, qkvwh, qkv_b, qkvh);

  // 4) head split; V transposed for contiguous B-fragments
  reshape_qkv_kernel<<<(unsigned)(((size_t)MM * QKVN) / 256), 256, 0, stream>>>(
      qkvh, Qh, Kh, Vth);

  // 5) flash attention with DPB bias (WMMA)
  attn_kernel<<<dim3(NN / 128, BB * HH), 256, 0, stream>>>(Qh, Kh, Vth, valsT, oh);

  // 6) output projection (WMMA): [4096,768] x [768,768] + b -> f32 d_out
  gemm_f16_kernel<DD, DD, true><<<dim3(DD / 128, MM / 128), 256, 0, stream>>>(
      oh, outwh, out_b, (float*)d_out);
}
